// SSIM_35545149342378
// MI455X (gfx1250) — matrix operand order repaired and checked by reading, not compiled, not run on hardware
//
#include <hip/hip_runtime.h>

typedef __attribute__((ext_vector_type(16))) _Float16 v16h;
typedef __attribute__((ext_vector_type(8)))  _Float16 v8h;
typedef __attribute__((ext_vector_type(8)))  float    v8f;

#define H_IMG 512
#define W_IMG 512
#define PLANES 48           // 16 * 3
#define TILE 64             // output tile per workgroup
#define RAW 80              // TILE + 8 halo each side (aligned for b128 LDS loads)
#define NBLOCKS (8*8*PLANES)

// Normalized 11-tap Gaussian, sigma = 1.5
__device__ __constant__ float GK[11] = {
    0.0010284f, 0.0075988f, 0.0360007f, 0.1093606f, 0.2130055f,
    0.2660116f,
    0.2130055f, 0.1093606f, 0.0360007f, 0.0075988f, 0.0010284f};

// A-operand fragment (16x32 f16) from row-major LDS tile.
// Lane l<16: row M=l, K={0..7,16..23}; lane>=16: row M=l-16, K={8..15,24..31}.
__device__ __forceinline__ v16h load_afrag(const _Float16* lds, int rowBase,
                                           int colBase, int lane) {
  const int row = rowBase + (lane & 15);
  const int col = colBase + ((lane >> 4) & 1) * 8;
  const v8h lo = *(const v8h*)(lds + row * RAW + col);       // K base .. base+7
  const v8h hi = *(const v8h*)(lds + row * RAW + col + 16);  // K base+16 ..
  v16h a;
#pragma unroll
  for (int i = 0; i < 8; ++i) { a[i] = lo[i]; a[8 + i] = hi[i]; }
  return a;
}

// Separable 11x11 conv of a 16x16 output tile via 3 WMMAs:
// H = A(16x32 raw) x wfB(32x16 banded), twice (rows 0..15, 16..31),
// OUT = wfA(16x32 banded) x H(32x16).  H's C-layout is remapped to the
// vertical B-layout with a single lane-half swap (shfl_xor 16).
__device__ __forceinline__ v8f conv2d_tile(v16h a0, v16h a1, v16h wfA, v16h wfB,
                                           bool laneLo) {
  v8f z = {};
  v8f h0 = __builtin_amdgcn_wmma_f32_16x16x32_f16(false, a0, false, wfB,
                                                  (short)0, z, false, false);
  v8f h1 = __builtin_amdgcn_wmma_f32_16x16x32_f16(false, a1, false, wfB,
                                                  (short)0, z, false, false);
  v16h hb;
#pragma unroll
  for (int i = 0; i < 8; ++i) {
    _Float16 c0 = (_Float16)h0[i];
    _Float16 c1 = (_Float16)h1[i];
    unsigned int pk =
        ((unsigned int)__builtin_bit_cast(unsigned short, c1) << 16) |
        (unsigned int)__builtin_bit_cast(unsigned short, c0);
    unsigned int sp = (unsigned int)__shfl_xor((int)pk, 16, 32);
    _Float16 s0 = __builtin_bit_cast(_Float16, (unsigned short)(sp & 0xffffu));
    _Float16 s1 = __builtin_bit_cast(_Float16, (unsigned short)(sp >> 16));
    hb[i]     = laneLo ? c0 : s1;  // K = i      (lanes<16) / 16+i (lanes>=16)
    hb[8 + i] = laneLo ? s0 : c1;  // K = 8+i    (lanes<16) / 24+i (lanes>=16)
  }
  return __builtin_amdgcn_wmma_f32_16x16x32_f16(false, wfA, false, hb,
                                                (short)0, z, false, false);
}

__global__ __launch_bounds__(256) void ssim_tile_kernel(
    const float* __restrict__ x, const float* __restrict__ y,
    float* __restrict__ partial) {
  __shared__ _Float16 sx[RAW * RAW];
  __shared__ _Float16 sy[RAW * RAW];
  __shared__ float sred[8];

  const int tid = threadIdx.x;
  const int lane = tid & 31;
  const int wave = tid >> 5;
  const bool laneLo = (lane < 16);

  const int bx = blockIdx.x, by = blockIdx.y, p = blockIdx.z;
  const float* xp = x + (size_t)p * (H_IMG * W_IMG);
  const float* yp = y + (size_t)p * (H_IMG * W_IMG);
  const int gx0 = bx * TILE - 8;
  const int gy0 = by * TILE - 8;

  // Stage 80x80 halo region of both images into LDS as f16 (zero-pad OOB,
  // matching SAME padding of the reference conv).
  for (int idx = tid; idx < RAW * RAW; idx += 256) {
    int r = idx / RAW;
    int c = idx - r * RAW;
    int gy = gy0 + r, gx = gx0 + c;
    float vx = 0.f, vy = 0.f;
    if (gy >= 0 && gy < H_IMG && gx >= 0 && gx < W_IMG) {
      int o = gy * W_IMG + gx;
      vx = xp[o];
      vy = yp[o];
    }
    sx[idx] = (_Float16)vx;
    sy[idx] = (_Float16)vy;
  }

  // Banded weight fragments (window shifted by -8, so band offset is +3:
  // W[o][k] = g[k - o - 3]).  wfA = A-operand layout, wfB = B-operand layout.
  const int half = (lane >> 4) & 1;
  const int mrow = lane & 15;
  v16h wfA, wfB;
#pragma unroll
  for (int i = 0; i < 16; ++i) {
    int kA = i + 16 * half;
    int dA = kA - mrow - 3;
    wfA[i] = (dA >= 0 && dA < 11) ? (_Float16)GK[dA] : (_Float16)0.f;
    int kB = i + 8 * half + ((i < 8) ? 0 : 8);
    int dB = kB - mrow - 3;
    wfB[i] = (dB >= 0 && dB < 11) ? (_Float16)GK[dB] : (_Float16)0.f;
  }

  __syncthreads();

  const float C1 = 1e-4f, C2 = 9e-4f;
  float acc = 0.f;

  // 16 tiles of 16x16 per block, 8 waves -> 2 tiles per wave.
  for (int tt = 0; tt < 2; ++tt) {
    const int t = wave * 2 + tt;
    const int rb = (t >> 2) * 16;  // LDS row base of 32-row vertical window
    const int cb = (t & 3) * 16;   // LDS col base of 32-col horizontal window

    v16h ax0 = load_afrag(sx, rb, cb, lane);
    v16h ax1 = load_afrag(sx, rb + 16, cb, lane);
    v16h ay0 = load_afrag(sy, rb, cb, lane);
    v16h ay1 = load_afrag(sy, rb + 16, cb, lane);

    v8f mux = conv2d_tile(ax0, ax1, wfA, wfB, laneLo);
    v8f muy = conv2d_tile(ay0, ay1, wfA, wfB, laneLo);
    v8f fxx = conv2d_tile(ax0 * ax0, ax1 * ax1, wfA, wfB, laneLo);
    v8f fyy = conv2d_tile(ay0 * ay0, ay1 * ay1, wfA, wfB, laneLo);
    v8f fxy = conv2d_tile(ax0 * ay0, ax1 * ay1, wfA, wfB, laneLo);

#pragma unroll
    for (int i = 0; i < 8; ++i) {
      float mx = mux[i], my = muy[i];
      float sxx = fxx[i] - mx * mx;
      float syy = fyy[i] - my * my;
      float sxy = fxy[i] - mx * my;
      float num = (2.f * mx * my + C1) * (2.f * sxy + C2);
      float den = (mx * mx + my * my + C1) * (sxx + syy + C2);
      acc += num / (den + 1e-12f);
    }
  }

  // wave32 reduction, then cross-wave via LDS; one partial per block.
#pragma unroll
  for (int off = 16; off > 0; off >>= 1) acc += __shfl_xor(acc, off, 32);
  if (lane == 0) sred[wave] = acc;
  __syncthreads();
  if (tid == 0) {
    float t = 0.f;
#pragma unroll
    for (int w = 0; w < 8; ++w) t += sred[w];
    partial[(blockIdx.z * gridDim.y + blockIdx.y) * gridDim.x + blockIdx.x] = t;
  }
}

__global__ __launch_bounds__(256) void ssim_reduce_kernel(
    const float* __restrict__ partial, int n, float* __restrict__ out) {
  __shared__ float sred[8];
  float acc = 0.f;
  for (int i = threadIdx.x; i < n; i += 256) acc += partial[i];
#pragma unroll
  for (int off = 16; off > 0; off >>= 1) acc += __shfl_xor(acc, off, 32);
  if ((threadIdx.x & 31) == 0) sred[threadIdx.x >> 5] = acc;
  __syncthreads();
  if (threadIdx.x == 0) {
    float t = 0.f;
#pragma unroll
    for (int w = 0; w < 8; ++w) t += sred[w];
    out[0] = t * (1.0f / (16.0f * 3.0f * 512.0f * 512.0f));
  }
}

extern "C" void kernel_launch(void* const* d_in, const int* in_sizes, int n_in,
                              void* d_out, int out_size, void* d_ws,
                              size_t ws_size, hipStream_t stream) {
  const float* x = (const float*)d_in[0];
  const float* y = (const float*)d_in[1];
  float* out = (float*)d_out;
  float* partial = (float*)d_ws;  // needs NBLOCKS floats = 12 KB

  dim3 grid(W_IMG / TILE, H_IMG / TILE, PLANES);  // 8 x 8 x 48 = 3072 blocks
  ssim_tile_kernel<<<grid, 256, 0, stream>>>(x, y, partial);
  ssim_reduce_kernel<<<1, 256, 0, stream>>>(partial, NBLOCKS, out);
}
